// TwoStageAttention_4140348474043
// MI455X (gfx1250) — matrix operand order repaired and checked by reading, not compiled, not run on hardware
//
#include <hip/hip_runtime.h>
#include <cstdint>
#include <cstddef>

// Problem constants (match reference)
#define BB   2
#define NNODE 10000
#define HH   128
#define EE   128000
#define TAUF 0.1f
#define ROWS (BB * EE)          // 256000 edge-rows
#define WAVES_PER_WG 4
#define LOG2E 1.4426950408889634f
// Per-wave LDS slots (ushort offsets), 16 rows each
#define SX0 0                   // 16x128
#define SX1 2048                // 16x128
#define SY0 4096                // 16x256  (f0 | b0)
#define SY1 8192                // 16x256  (f1 | b1)
#define SZ0 12288               // 16x256  (f0' | b0')
#define LDS_PER_WAVE 16384      // ushorts = 32KB

// Packed-weight offsets in workspace (ushort elements)
#define WIH_L0F 0
#define WHH_L0F 49152
#define WIH_L0B 98304
#define WHH_L0B 147456
#define WIH_L1F 196608
#define WHH_L1F 294912
#define WIH_L1B 344064
#define WHH_L1B 442368
#define PACKED_TOTAL 491520     // ushorts

typedef __attribute__((ext_vector_type(16))) __bf16 v16bf;
typedef __attribute__((ext_vector_type(8)))  float  v8f;

union Frag {
    v16bf v;
    uint4 q[2];
    unsigned short us[16];
};

__device__ __forceinline__ unsigned short f2bf(float f) {
    union { float f; unsigned int u; } c; c.f = f;
    unsigned int u = c.u;
    unsigned int r = u + 0x7FFFu + ((u >> 16) & 1u);   // round-to-nearest-even
    return (unsigned short)(r >> 16);
}
__device__ __forceinline__ float bf2f(unsigned short b) {
    union { float f; unsigned int u; } c; c.u = ((unsigned int)b) << 16; return c.f;
}

// Fast gate activations: native v_exp_f32 (2^x) + v_rcp_f32 on the TRANS pipe,
// so the VALU tail stays short and co-executes with the WMMA pipe.
__device__ __forceinline__ float fsigmoid(float x) {
    return __builtin_amdgcn_rcpf(1.f + __builtin_amdgcn_exp2f(-x * LOG2E));
}
__device__ __forceinline__ float ftanh(float x) {
    return 1.f - 2.f * __builtin_amdgcn_rcpf(__builtin_amdgcn_exp2f((2.f * LOG2E) * x) + 1.f);
}

__device__ __forceinline__ v8f wmma_bf16(const Frag& a, const Frag& b, v8f c) {
    // 8 args: (neg_a, A, neg_b, B, c_mod, C, reuse_a, reuse_b)
    return __builtin_amdgcn_wmma_f32_16x16x32_bf16(false, a.v, false, b.v,
                                                   (short)0, c, false, false);
}

// A fragment (16x32 bf16, MxK) from LDS per ISA 16-bit A layout:
// lane<16 : M=lane,    elems 0..7 -> K=kb+0..7,   elems 8..15 -> K=kb+16..23
// lane>=16: M=lane-16, elems 0..7 -> K=kb+8..15,  elems 8..15 -> K=kb+24..31
__device__ __forceinline__ Frag load_a(const unsigned short* lds, int base,
                                       int stride, int kbase, int lane) {
    const int row = lane & 15;
    const int hi  = lane >> 4;
    const int o   = base + row * stride + kbase + hi * 8;
    Frag f;
    f.q[0] = *reinterpret_cast<const uint4*>(lds + o);        // ds_load_b128
    f.q[1] = *reinterpret_cast<const uint4*>(lds + o + 16);   // ds_load_b128
    return f;
}

// B fragment from pre-packed weights: 16 contiguous bf16 per lane.
__device__ __forceinline__ Frag load_b(const unsigned short* __restrict__ pw,
                                       size_t base) {
    const uint4* p = reinterpret_cast<const uint4*>(pw + base);
    Frag f;
    f.q[0] = p[0];    // global_load_b128
    f.q[1] = p[1];
    return f;
}

__device__ __forceinline__ size_t bofs(int nt, int nk, int kt, int lane) {
    return ((size_t)((nt * nk + kt) * 32 + lane)) * 16;
}

// One GRU cell on a 16-row tile. x: LDS bf16 [16 x DIN]; hp: LDS bf16 [16 x 128]
// (or absent). Output h_new -> LDS bf16 [16 x 128] at out_off with out_stride.
// A fragments are hoisted into registers once per cell (reused by all 8 j-tiles
// and all 3 gates) -- the compiler cannot hoist these itself because the LDS
// output writes may alias the x slot as far as it can prove.
template<int DIN, bool HAS_HP>
__device__ __forceinline__ void gru_cell_tile(
    unsigned short* lds,
    int x_off, int x_stride,
    int hp_off, int hp_stride,
    const unsigned short* __restrict__ wih,
    const unsigned short* __restrict__ whh,
    const float* __restrict__ b_ih, const float* __restrict__ b_hh,
    int out_off, int out_stride, int lane)
{
    constexpr int NKX = DIN / 32;
    constexpr int NKH = HH / 32;     // 4
    const int hi   = lane >> 4;
    const int ncol = lane & 15;
    const v8f z8 = {0.f, 0.f, 0.f, 0.f, 0.f, 0.f, 0.f, 0.f};

    Frag ax[NKX];
    #pragma unroll
    for (int kt = 0; kt < NKX; ++kt)
        ax[kt] = load_a(lds, x_off, x_stride, kt * 32, lane);
    Frag ah[NKH];
    if constexpr (HAS_HP) {
        #pragma unroll
        for (int kt = 0; kt < NKH; ++kt)
            ah[kt] = load_a(lds, hp_off, hp_stride, kt * 32, lane);
    }

    for (int j = 0; j < 8; ++j) {    // 8 N-tiles of 16 cover H=128 per gate
        v8f acc_r = z8, acc_z = z8, acc_in = z8, acc_hn = z8;
        #pragma unroll
        for (int kt = 0; kt < NKX; ++kt) {
            acc_r  = wmma_bf16(ax[kt], load_b(wih, bofs(j,      NKX, kt, lane)), acc_r);
            acc_z  = wmma_bf16(ax[kt], load_b(wih, bofs(8 + j,  NKX, kt, lane)), acc_z);
            acc_in = wmma_bf16(ax[kt], load_b(wih, bofs(16 + j, NKX, kt, lane)), acc_in);
        }
        if constexpr (HAS_HP) {
            #pragma unroll
            for (int kt = 0; kt < NKH; ++kt) {
                acc_r  = wmma_bf16(ah[kt], load_b(whh, bofs(j,      NKH, kt, lane)), acc_r);
                acc_z  = wmma_bf16(ah[kt], load_b(whh, bofs(8 + j,  NKH, kt, lane)), acc_z);
                acc_hn = wmma_bf16(ah[kt], load_b(whh, bofs(16 + j, NKH, kt, lane)), acc_hn);
            }
        }
        const int feat = j * 16 + ncol;
        const float br2 = b_ih[feat]       + b_hh[feat];
        const float bz2 = b_ih[128 + feat] + b_hh[128 + feat];
        const float bin = b_ih[256 + feat];
        const float bhn = b_hh[256 + feat];
        #pragma unroll
        for (int v = 0; v < 8; ++v) {
            const int M = v + hi * 8;          // C/D layout: VGPR v, half-lane hi
            float r  = fsigmoid(acc_r[v] + br2);
            float zz = fsigmoid(acc_z[v] + bz2);
            float hn = (HAS_HP ? acc_hn[v] : 0.f) + bhn;
            float nn = ftanh(acc_in[v] + bin + r * hn);
            float hp = HAS_HP ? bf2f(lds[hp_off + M * hp_stride + feat]) : 0.f;
            lds[out_off + M * out_stride + feat] = f2bf((1.f - zz) * nn + zz * hp);
        }
    }
}

// ---------------- Fused gather + BiGRU(2 layers) + gelu-logit ----------------
__global__ void __launch_bounds__(WAVES_PER_WG * 32)
gru_logits_kernel(const float* __restrict__ h, const int* __restrict__ edge_index,
                  const unsigned short* __restrict__ pw,
                  const float* __restrict__ bi0f, const float* __restrict__ bh0f,
                  const float* __restrict__ bi0b, const float* __restrict__ bh0b,
                  const float* __restrict__ bi1f, const float* __restrict__ bh1f,
                  const float* __restrict__ bi1b, const float* __restrict__ bh1b,
                  const float* __restrict__ w_con, const float* __restrict__ b_con,
                  float* __restrict__ lt)
{
    extern __shared__ unsigned short smem[];
    const int lane = threadIdx.x & 31;
    const int wave = threadIdx.x >> 5;
    unsigned short* lds = smem + wave * LDS_PER_WAVE;

    const int tile = blockIdx.x * WAVES_PER_WG + wave;
    const int row0 = tile * 16;

    // Stage x0 (h_target) and x1 (h_source) as bf16 rows in LDS.
    const int c = lane * 4;
    for (int r = 0; r < 16; ++r) {
        const int rowi = row0 + r;
        const int b = rowi / EE;
        const int e = rowi - b * EE;
        const int tg = edge_index[e];           // row 0 = tgt
        const int sr = edge_index[EE + e];      // row 1 = src
        const float4 a0 = *reinterpret_cast<const float4*>(
            h + ((size_t)b * NNODE + tg) * HH + c);
        const float4 a1 = *reinterpret_cast<const float4*>(
            h + ((size_t)b * NNODE + sr) * HH + c);
        uint2 p0, p1;
        p0.x = (unsigned)f2bf(a0.x) | ((unsigned)f2bf(a0.y) << 16);
        p0.y = (unsigned)f2bf(a0.z) | ((unsigned)f2bf(a0.w) << 16);
        p1.x = (unsigned)f2bf(a1.x) | ((unsigned)f2bf(a1.y) << 16);
        p1.y = (unsigned)f2bf(a1.z) | ((unsigned)f2bf(a1.w) << 16);
        *reinterpret_cast<uint2*>(lds + SX0 + r * 128 + c) = p0;
        *reinterpret_cast<uint2*>(lds + SX1 + r * 128 + c) = p1;
    }

    // Layer 0 (din = 128): f0, f1, b1, b0
    gru_cell_tile<128, false>(lds, SX0, 128, 0, 0,           pw + WIH_L0F, pw + WHH_L0F, bi0f, bh0f, SY0 + 0,   256, lane);
    gru_cell_tile<128, true >(lds, SX1, 128, SY0 + 0, 256,   pw + WIH_L0F, pw + WHH_L0F, bi0f, bh0f, SY1 + 0,   256, lane);
    gru_cell_tile<128, false>(lds, SX1, 128, 0, 0,           pw + WIH_L0B, pw + WHH_L0B, bi0b, bh0b, SY1 + 128, 256, lane);
    gru_cell_tile<128, true >(lds, SX0, 128, SY1 + 128, 256, pw + WIH_L0B, pw + WHH_L0B, bi0b, bh0b, SY0 + 128, 256, lane);
    // Layer 1 (din = 256): f0', b1' (scratch in SX0), b0'   (f1' is dead)
    gru_cell_tile<256, false>(lds, SY0, 256, 0, 0,           pw + WIH_L1F, pw + WHH_L1F, bi1f, bh1f, SZ0 + 0,   256, lane);
    gru_cell_tile<256, false>(lds, SY1, 256, 0, 0,           pw + WIH_L1B, pw + WHH_L1B, bi1b, bh1b, SX0 + 0,   128, lane);
    gru_cell_tile<256, true >(lds, SY0, 256, SX0, 128,       pw + WIH_L1B, pw + WHH_L1B, bi1b, bh1b, SZ0 + 128, 256, lane);

    // logit = gelu(z0 . w_con + b_con): two lanes per row, vectorized LDS reads.
    const int m    = lane >> 1;
    const int half = lane & 1;
    const int k0   = half * 128;
    float s = 0.f;
    #pragma unroll
    for (int kk = 0; kk < 128; kk += 8) {
        const uint4 q = *reinterpret_cast<const uint4*>(lds + SZ0 + m * 256 + k0 + kk);
        const unsigned short* u = reinterpret_cast<const unsigned short*>(&q);
        const float4 w0 = *reinterpret_cast<const float4*>(w_con + k0 + kk);
        const float4 w1 = *reinterpret_cast<const float4*>(w_con + k0 + kk + 4);
        s += bf2f(u[0]) * w0.x + bf2f(u[1]) * w0.y + bf2f(u[2]) * w0.z + bf2f(u[3]) * w0.w;
        s += bf2f(u[4]) * w1.x + bf2f(u[5]) * w1.y + bf2f(u[6]) * w1.z + bf2f(u[7]) * w1.w;
    }
    s += __shfl_xor(s, 1, 32);
    if (half == 0) {
        const float x = s + b_con[0];
        const float g = 0.5f * x * (1.f + erff(x * 0.70710678118654752f));
        const int rowi = row0 + m;
        const int b = rowi / EE;
        const int e = rowi - b * EE;
        lt[(size_t)e * BB + b] = g;   // lT layout (E, B)
    }
}

// ---------------- Weight packing: f32 (3H x din) -> WMMA-B bf16 fragments ----
__global__ void pack_w_kernel(const float* __restrict__ w,
                              unsigned short* __restrict__ out, int din)
{
    const int nk = din >> 5;
    const int total = 24 * nk * 512;
    const int idx = blockIdx.x * blockDim.x + threadIdx.x;
    if (idx >= total) return;
    const int i    = idx & 15;
    const int lane = (idx >> 4) & 31;
    const int t    = idx >> 9;           // nt*nk + kt
    const int kt   = t % nk;
    const int nt   = t / nk;
    const int k    = kt * 32 + ((lane >> 4) ? 8 : 0) + i + ((i >= 8) ? 8 : 0);
    const int n    = nt * 16 + (lane & 15);
    out[idx] = f2bf(w[(size_t)n * din + k]);
}

// ---------------- Init ----------------
__global__ void init_kernel(float* __restrict__ out, int out_size,
                            unsigned int* __restrict__ mh, unsigned int* __restrict__ ms,
                            float* __restrict__ sh, float* __restrict__ ss, int nb)
{
    const int idx = blockIdx.x * blockDim.x + threadIdx.x;
    if (idx < out_size) out[idx] = 0.f;
    if (idx < nb) { mh[idx] = 0u; ms[idx] = 0u; sh[idx] = 0.f; ss[idx] = 0.f; }
}

// Monotone int key for float atomic max
__device__ __forceinline__ unsigned int fkey(float f) {
    union { float f; unsigned int u; } c; c.f = f;
    return (c.u & 0x80000000u) ? ~c.u : (c.u | 0x80000000u);
}
__device__ __forceinline__ float fval(unsigned int k) {
    union { float f; unsigned int u; } c;
    c.u = (k & 0x80000000u) ? (k & 0x7FFFFFFFu) : ~k;
    return c.f;
}
__device__ __forceinline__ float fexp(float x) {
    return __builtin_amdgcn_exp2f(x * LOG2E);
}
__device__ __forceinline__ float gumbel_noise(unsigned int e, unsigned int b) {
    unsigned int s = e * 0x9E3779B1u + b * 0x85EBCA77u + 0x1234567u;
    s ^= s >> 16; s *= 0x7FEB352Du; s ^= s >> 15; s *= 0x846CA68Bu; s ^= s >> 16;
    const float u = ((float)((s >> 8) + 1u)) * (1.0f / 16777218.0f);
    return -logf(-logf(u));
}

// ---------------- Segmented softmax passes ----------------
__global__ void seg_max_kernel(const float* __restrict__ lt,
                               const int* __restrict__ edge_index,
                               unsigned int* __restrict__ mh,
                               unsigned int* __restrict__ ms)
{
    const int idx = blockIdx.x * blockDim.x + threadIdx.x;
    if (idx >= EE * BB) return;
    const int e = idx / BB, b = idx - e * BB;
    const int src = edge_index[EE + e];
    const float l = lt[idx];
    const float xh = (l + gumbel_noise(e, b)) * (1.0f / TAUF);
    atomicMax(&mh[src * BB + b], fkey(xh));
    atomicMax(&ms[src * BB + b], fkey(l));
}

__global__ void seg_sum_kernel(const float* __restrict__ lt,
                               const int* __restrict__ edge_index,
                               const unsigned int* __restrict__ mh,
                               const unsigned int* __restrict__ ms,
                               float* __restrict__ sh, float* __restrict__ ss)
{
    const int idx = blockIdx.x * blockDim.x + threadIdx.x;
    if (idx >= EE * BB) return;
    const int e = idx / BB, b = idx - e * BB;
    const int src = edge_index[EE + e];
    const float l = lt[idx];
    const float xh = (l + gumbel_noise(e, b)) * (1.0f / TAUF);
    atomicAdd(&sh[src * BB + b], fexp(xh - fval(mh[src * BB + b])));
    atomicAdd(&ss[src * BB + b], fexp(l  - fval(ms[src * BB + b])));
}

// ---------------- Weighted scatter-add into output ----------------
__global__ void scatter_kernel(const float* __restrict__ h,
                               const int* __restrict__ edge_index,
                               const float* __restrict__ lt,
                               const unsigned int* __restrict__ mh,
                               const unsigned int* __restrict__ ms,
                               const float* __restrict__ sh,
                               const float* __restrict__ ss,
                               float* __restrict__ out)
{
    const int idx = blockIdx.x * blockDim.x + threadIdx.x;
    if (idx >= EE * BB * (HH / 4)) return;
    const int hq = idx & 31;             // HH/4 = 32 chunks of float4
    const int eb = idx >> 5;
    const int e = eb / BB, b = eb - e * BB;
    const int src = edge_index[EE + e];
    const int tgt = edge_index[e];
    const float l  = lt[eb];
    const float xh = (l + gumbel_noise(e, b)) * (1.0f / TAUF);
    const int sb = src * BB + b;
    const float hard = fexp(xh - fval(mh[sb])) / sh[sb];
    const float soft = fexp(l  - fval(ms[sb])) / ss[sb];
    const float w = hard * soft;
    const float4 v = *reinterpret_cast<const float4*>(
        h + ((size_t)b * NNODE + src) * HH + hq * 4);
    float* op = out + ((size_t)b * NNODE + tgt) * HH + hq * 4;
    atomicAdd(op + 0, v.x * w);
    atomicAdd(op + 1, v.y * w);
    atomicAdd(op + 2, v.z * w);
    atomicAdd(op + 3, v.w * w);
}

static inline unsigned cdiv(unsigned a, unsigned b) { return (a + b - 1) / b; }

extern "C" void kernel_launch(void* const* d_in, const int* in_sizes, int n_in,
                              void* d_out, int out_size, void* d_ws, size_t ws_size,
                              hipStream_t stream)
{
    (void)in_sizes; (void)n_in; (void)ws_size;
    const float* h        = (const float*)d_in[0];
    const int*   ei       = (const int*)d_in[1];
    const float* w_ih_l0f = (const float*)d_in[2];
    const float* w_hh_l0f = (const float*)d_in[3];
    const float* b_ih_l0f = (const float*)d_in[4];
    const float* b_hh_l0f = (const float*)d_in[5];
    const float* w_ih_l0b = (const float*)d_in[6];
    const float* w_hh_l0b = (const float*)d_in[7];
    const float* b_ih_l0b = (const float*)d_in[8];
    const float* b_hh_l0b = (const float*)d_in[9];
    const float* w_ih_l1f = (const float*)d_in[10];
    const float* w_hh_l1f = (const float*)d_in[11];
    const float* b_ih_l1f = (const float*)d_in[12];
    const float* b_hh_l1f = (const float*)d_in[13];
    const float* w_ih_l1b = (const float*)d_in[14];
    const float* w_hh_l1b = (const float*)d_in[15];
    const float* b_ih_l1b = (const float*)d_in[16];
    const float* b_hh_l1b = (const float*)d_in[17];
    const float* w_con    = (const float*)d_in[18];
    const float* b_con    = (const float*)d_in[19];
    float* out = (float*)d_out;

    // Workspace carve-up
    char* ws = (char*)d_ws;
    unsigned short* pw = (unsigned short*)ws;               // PACKED_TOTAL ushorts
    size_t off = (size_t)PACKED_TOTAL * 2;                  // 983040 B (16B aligned)
    float* lt = (float*)(ws + off);        off += (size_t)EE * BB * 4;
    unsigned int* mh = (unsigned int*)(ws + off); off += (size_t)NNODE * BB * 4;
    unsigned int* ms = (unsigned int*)(ws + off); off += (size_t)NNODE * BB * 4;
    float* sh = (float*)(ws + off);        off += (size_t)NNODE * BB * 4;
    float* ss = (float*)(ws + off);        off += (size_t)NNODE * BB * 4;

    // Init output + segment buffers
    {
        const unsigned total = (unsigned)(out_size > NNODE * BB ? out_size : NNODE * BB);
        init_kernel<<<cdiv(total, 256), 256, 0, stream>>>(out, out_size, mh, ms, sh, ss, NNODE * BB);
    }
    // Pack the 8 weight matrices into WMMA-B bf16 fragment order
    pack_w_kernel<<<cdiv(49152, 256), 256, 0, stream>>>(w_ih_l0f, pw + WIH_L0F, 128);
    pack_w_kernel<<<cdiv(49152, 256), 256, 0, stream>>>(w_hh_l0f, pw + WHH_L0F, 128);
    pack_w_kernel<<<cdiv(49152, 256), 256, 0, stream>>>(w_ih_l0b, pw + WIH_L0B, 128);
    pack_w_kernel<<<cdiv(49152, 256), 256, 0, stream>>>(w_hh_l0b, pw + WHH_L0B, 128);
    pack_w_kernel<<<cdiv(98304, 256), 256, 0, stream>>>(w_ih_l1f, pw + WIH_L1F, 256);
    pack_w_kernel<<<cdiv(49152, 256), 256, 0, stream>>>(w_hh_l1f, pw + WHH_L1F, 128);
    pack_w_kernel<<<cdiv(98304, 256), 256, 0, stream>>>(w_ih_l1b, pw + WIH_L1B, 256);
    pack_w_kernel<<<cdiv(49152, 256), 256, 0, stream>>>(w_hh_l1b, pw + WHH_L1B, 128);

    // Fused BiGRU + logits: 16 rows/wave, 4 waves/WG -> 4000 WGs, 128KB dyn LDS
    {
        const unsigned tiles  = ROWS / 16;                  // 16000
        const unsigned blocks = tiles / WAVES_PER_WG;       // 4000
        const size_t   shmem  = (size_t)WAVES_PER_WG * LDS_PER_WAVE * 2;  // 131072 B
        gru_logits_kernel<<<blocks, WAVES_PER_WG * 32, shmem, stream>>>(
            h, ei, pw,
            b_ih_l0f, b_hh_l0f, b_ih_l0b, b_hh_l0b,
            b_ih_l1f, b_hh_l1f, b_ih_l1b, b_hh_l1b,
            w_con, b_con, lt);
    }

    // Segmented softmax (hard gumbel + soft) and weighted scatter-add
    seg_max_kernel<<<cdiv(EE * BB, 256), 256, 0, stream>>>(lt, ei, mh, ms);
    seg_sum_kernel<<<cdiv(EE * BB, 256), 256, 0, stream>>>(lt, ei, mh, ms, sh, ss);
    scatter_kernel<<<cdiv(EE * BB * (HH / 4), 256), 256, 0, stream>>>(
        h, ei, lt, mh, ms, sh, ss, out);
}
